// Graph_Transformer_13211319402621
// MI455X (gfx1250) — matrix operand order repaired and checked by reading, not compiled, not run on hardware
//
#include <hip/hip_runtime.h>
#include <stdint.h>

// Problem constants (match reference)
#define N_NODES 100000
#define N_EDGES 1600000
#define DIM     128     // IN == H*C == 128 for every layer
#define HEADS   4
#define CH      32
#define LAYERS  3
#define RT_PER_BLOCK 5  // row tiles per GEMM block (6250 / 5 = 1250 blocks)

typedef __attribute__((ext_vector_type(16))) __bf16 v16bf;
typedef __attribute__((ext_vector_type(8)))  float  v8f;

// ---------- helpers ----------
__device__ __forceinline__ unsigned short f2bf(float f) {
  unsigned u = __float_as_uint(f);
  unsigned r = u + 0x7FFFu + ((u >> 16) & 1u);   // round-to-nearest-even
  return (unsigned short)(r >> 16);
}

// Monotonic float -> uint key (order preserving), for atomicMax on floats.
__device__ __forceinline__ unsigned fkey(float f) {
  unsigned b = __float_as_uint(f);
  return b ^ ((unsigned)(((int)b) >> 31) | 0x80000000u);
}
__device__ __forceinline__ float funkey(unsigned k) {
  unsigned b = (k & 0x80000000u) ? (k ^ 0x80000000u) : ~k;
  return __uint_as_float(b);
}

// ---------- kernel 1: fp32 -> bf16 pack (4 elems/thread) ----------
__global__ void cvt_f32_bf16_kernel(const float* __restrict__ in,
                                    unsigned short* __restrict__ out, int n4) {
  int i = blockIdx.x * blockDim.x + threadIdx.x;
  if (i >= n4) return;
  float4 v = ((const float4*)in)[i];
  unsigned lo = (unsigned)f2bf(v.x) | ((unsigned)f2bf(v.y) << 16);
  unsigned hi = (unsigned)f2bf(v.z) | ((unsigned)f2bf(v.w) << 16);
  ((uint2*)out)[i] = make_uint2(lo, hi);
}

// ---------- kernel 2: swizzle one 128x128 fp32 weight into WMMA B-fragment
// layout, bf16. Layout: frag(kstep 0..3, ctile 0..7) = 32 lanes x 16 ushorts.
// Lane (half,n): element i = W[kstep*32 + half*16 + i][ctile*16 + n].
__global__ void swizzle_w_kernel(const float* __restrict__ W,
                                 unsigned short* __restrict__ swz) {
  int s = blockIdx.x * blockDim.x + threadIdx.x;   // 0..16383
  if (s >= 4 * 8 * 32 * 16) return;
  int i     = s & 15;
  int lane  = (s >> 4) & 31;
  int ctile = (s >> 9) & 7;
  int kstep = (s >> 12) & 3;
  int n = lane & 15, half = lane >> 4;
  int k   = kstep * 32 + half * 16 + i;
  int col = ctile * 16 + n;
  swz[s] = f2bf(W[k * DIM + col]);
}

// ---------- kernel 3: GEMM  O[N x 128] = A(bf16) @ Wswz(bf16) + bias ----------
// Block = 128 threads (4 waves), RT_PER_BLOCK row tiles per block.
// W is staged global->LDS once via async-copy (ASYNCcnt), then each wave pulls
// its B fragments from LDS into registers once and reuses them for all tiles.
// Steady state per tile: 8 coalesced A loads + 8 WMMAs + 16 imm-offset stores.
__global__ void __launch_bounds__(128)
gemm_bf16_kernel(const unsigned short* __restrict__ A,
                 const unsigned short* __restrict__ Wswz,
                 const float* __restrict__ bias,
                 float* __restrict__ O) {
  __shared__ unsigned short wlds[16384];           // 32KB swizzled bf16 weight

  const int tid  = threadIdx.x;
  const int lane = tid & 31;
  const int wave = tid >> 5;

  // ---- one-time async stage: global (L2-resident) -> LDS, 16B per op ----
  {
    unsigned lbase = (unsigned)(uintptr_t)(void*)wlds;   // LDS offset (addr[31:0])
    unsigned long long gbase = (unsigned long long)(uintptr_t)Wswz;
#pragma unroll
    for (int i = 0; i < 16; ++i) {
      unsigned loff = lbase + (unsigned)(tid + i * 128) * 16u;
      unsigned long long ga = gbase + (unsigned long long)(tid + i * 128) * 16ull;
      asm volatile("global_load_async_to_lds_b128 %0, %1, off"
                   :: "v"(loff), "v"(ga) : "memory");
    }
    asm volatile("s_wait_asynccnt 0x0" ::: "memory");
  }
  __syncthreads();

  const int n   = lane & 15;      // A row within tile AND D column within tile
  const int hi  = lane >> 4;
  const int ct0 = wave * 2;       // wave owns ctiles {ct0, ct0+1}

  union Frag { uint4 u[2]; v16bf b; };

  // ---- load all B fragments into registers once (ds_load_b128 x16) ----
  Frag B0[4], B1[4];
#pragma unroll
  for (int ks = 0; ks < 4; ++ks) {
    const unsigned short* p0 = &wlds[((ks * 8 + ct0) * 32 + lane) << 4];
    B0[ks].u[0] = *(const uint4*)(p0);
    B0[ks].u[1] = *(const uint4*)(p0 + 8);
    B1[ks].u[0] = *(const uint4*)(p0 + 512);       // next ctile = +32 lanes*16
    B1[ks].u[1] = *(const uint4*)(p0 + 520);
  }

  const float bb0 = bias[ct0 * 16 + n];
  const float bb1 = bias[ct0 * 16 + 16 + n];

  // ---- hoisted, affine base pointers (bump by constant stride per tile) ----
  const unsigned short* Arow = A + (size_t)(blockIdx.x * RT_PER_BLOCK * 16 + n) * DIM;
  float* ob = O + (size_t)(blockIdx.x * RT_PER_BLOCK * 16 + hi * 8) * DIM
                + ct0 * 16 + n;

  // ---- steady state: A loads + WMMA only ----
#pragma unroll
  for (int t = 0; t < RT_PER_BLOCK; ++t) {
    v8f acc0 = {}, acc1 = {};
#pragma unroll
    for (int ks = 0; ks < 4; ++ks) {
      Frag a;
      const int ka = ks * 32 + hi * 16;            // A: K-contiguous per lane
      a.u[0] = *(const uint4*)(Arow + ka);
      a.u[1] = *(const uint4*)(Arow + ka + 8);
      acc0 = __builtin_amdgcn_wmma_f32_16x16x32_bf16(false, a.b, false, B0[ks].b,
                                                     (short)0, acc0, false, false);
      acc1 = __builtin_amdgcn_wmma_f32_16x16x32_bf16(false, a.b, false, B1[ks].b,
                                                     (short)0, acc1, false, false);
    }
    // D layout: VGPR r -> row (hi*8 + r); affine -> immediate-offset stores
#pragma unroll
    for (int r = 0; r < 8; ++r) {
      ob[r * DIM]      = acc0[r] + bb0;
      ob[r * DIM + 16] = acc1[r] + bb1;
    }
    Arow += 16 * DIM;
    ob   += 16 * DIM;
  }
}

// ---------- kernel 4: init segment max keys / denom ----------
__global__ void init_md_kernel(unsigned* __restrict__ mkeys,
                               float* __restrict__ denom, int n) {
  int i = blockIdx.x * blockDim.x + threadIdx.x;
  if (i < n) { mkeys[i] = 0u; denom[i] = 0.0f; }
}

// ---------- kernel 5: per-(edge,head) logits + segment max ----------
__global__ void edge_logits_kernel(const int* __restrict__ src,
                                   const int* __restrict__ dst,
                                   const float* __restrict__ q,
                                   const float* __restrict__ k,
                                   float* __restrict__ logits,
                                   unsigned* __restrict__ mkeys) {
  int gid = blockIdx.x * blockDim.x + threadIdx.x;
  if (gid >= N_EDGES * HEADS) return;
  int e = gid >> 2, h = gid & 3;
  int s = src[e], d = dst[e];
  const float4* qp = (const float4*)(q + (size_t)d * DIM + h * CH);
  const float4* kp = (const float4*)(k + (size_t)s * DIM + h * CH);
  float acc = 0.0f;
#pragma unroll
  for (int i = 0; i < 8; ++i) {
    float4 a = qp[i], b = kp[i];
    acc += a.x * b.x + a.y * b.y + a.z * b.z + a.w * b.w;
  }
  acc *= 0.17677669529663687f;                     // 1/sqrt(32)
  logits[gid] = acc;
  atomicMax(&mkeys[d * HEADS + h], fkey(acc));
}

// ---------- kernel 6: ex = exp(logit - max); denom += ex ----------
__global__ void edge_exp_kernel(const int* __restrict__ dst,
                                const unsigned* __restrict__ mkeys,
                                float* __restrict__ logits,
                                float* __restrict__ denom) {
  int gid = blockIdx.x * blockDim.x + threadIdx.x;
  if (gid >= N_EDGES * HEADS) return;
  int e = gid >> 2, h = gid & 3;
  int d = dst[e];
  float m  = funkey(mkeys[d * HEADS + h]);
  float ex = __expf(logits[gid] - m);
  logits[gid] = ex;
  unsafeAtomicAdd(&denom[d * HEADS + h], ex);
}

// ---------- kernel 7: out[dst] += alpha * v[src]  (into skip buffer) ----------
__global__ void edge_scatter_kernel(const int* __restrict__ src,
                                    const int* __restrict__ dst,
                                    const float* __restrict__ ex,
                                    const float* __restrict__ denom,
                                    const float* __restrict__ v,
                                    float* __restrict__ out) {
  int gid = blockIdx.x * blockDim.x + threadIdx.x;
  if (gid >= N_EDGES * HEADS) return;
  int e = gid >> 2, h = gid & 3;
  int s = src[e], d = dst[e];
  float den = denom[d * HEADS + h];
  float alpha = ex[gid] / den;
  const float4* vp = (const float4*)(v + (size_t)s * DIM + h * CH);
  float* op = out + (size_t)d * DIM + h * CH;
#pragma unroll
  for (int i = 0; i < 8; ++i) {
    float4 vv = vp[i];
    unsafeAtomicAdd(op + 4 * i + 0, alpha * vv.x);
    unsafeAtomicAdd(op + 4 * i + 1, alpha * vv.y);
    unsafeAtomicAdd(op + 4 * i + 2, alpha * vv.z);
    unsafeAtomicAdd(op + 4 * i + 3, alpha * vv.w);
  }
}

// ---------- kernel 8: relu + repack bf16 (next layer A); optional fp32 out ---
__global__ void relu_cvt_kernel(const float* __restrict__ in,
                                unsigned short* __restrict__ hbf,
                                float* __restrict__ fout, int n4) {
  int i = blockIdx.x * blockDim.x + threadIdx.x;
  if (i >= n4) return;
  float4 v = ((const float4*)in)[i];
  v.x = fmaxf(v.x, 0.0f); v.y = fmaxf(v.y, 0.0f);
  v.z = fmaxf(v.z, 0.0f); v.w = fmaxf(v.w, 0.0f);
  unsigned lo = (unsigned)f2bf(v.x) | ((unsigned)f2bf(v.y) << 16);
  unsigned hi = (unsigned)f2bf(v.z) | ((unsigned)f2bf(v.w) << 16);
  ((uint2*)hbf)[i] = make_uint2(lo, hi);
  if (fout) ((float4*)fout)[i] = v;
}

// ---------------------------------------------------------------------------
extern "C" void kernel_launch(void* const* d_in, const int* in_sizes, int n_in,
                              void* d_out, int out_size, void* d_ws, size_t ws_size,
                              hipStream_t stream) {
  const float* x   = (const float*)d_in[0];
  const int*   ei  = (const int*)d_in[1];
  const int*   srcIdx = ei;              // edge_index[0]
  const int*   dstIdx = ei + N_EDGES;    // edge_index[1]
  // params: d_in[2 + l*8 + {0:Wq,1:bq,2:Wk,3:bk,4:Wv,5:bv,6:Ws,7:bs}]

  char* ws = (char*)d_ws;
  size_t off = 0;
  auto alloc = [&](size_t bytes) -> void* {
    void* p = ws + off;
    off = (off + bytes + 255) & ~(size_t)255;
    return p;
  };
  unsigned short* hbf   = (unsigned short*)alloc((size_t)N_NODES * DIM * 2);
  unsigned short* wswz  = (unsigned short*)alloc((size_t)LAYERS * 4 * 16384 * 2);
  float* qb    = (float*)alloc((size_t)N_NODES * DIM * 4);
  float* kb    = (float*)alloc((size_t)N_NODES * DIM * 4);
  float* vb    = (float*)alloc((size_t)N_NODES * DIM * 4);
  float* sb    = (float*)alloc((size_t)N_NODES * DIM * 4);
  float* exbuf = (float*)alloc((size_t)N_EDGES * HEADS * 4);
  unsigned* mkeys = (unsigned*)alloc((size_t)N_NODES * HEADS * 4);
  float*    denom = (float*)alloc((size_t)N_NODES * HEADS * 4);

  const int nElem4 = N_NODES * DIM / 4;            // 3.2M
  const int ehThreads = N_EDGES * HEADS;           // 6.4M
  const int gemmBlocks = N_NODES / (16 * RT_PER_BLOCK);   // 1250

  // x -> bf16 (layer-0 A matrix)
  cvt_f32_bf16_kernel<<<(nElem4 + 255) / 256, 256, 0, stream>>>(x, hbf, nElem4);

  // pre-swizzle all 12 weights into WMMA B-fragment layout (bf16, L2-resident)
  for (int l = 0; l < LAYERS; ++l)
    for (int w = 0; w < 4; ++w)
      swizzle_w_kernel<<<64, 256, 0, stream>>>(
          (const float*)d_in[2 + l * 8 + w * 2], wswz + (size_t)(l * 4 + w) * 16384);

  for (int l = 0; l < LAYERS; ++l) {
    const float* bqp = (const float*)d_in[2 + l * 8 + 1];
    const float* bkp = (const float*)d_in[2 + l * 8 + 3];
    const float* bvp = (const float*)d_in[2 + l * 8 + 5];
    const float* bsp = (const float*)d_in[2 + l * 8 + 7];
    const unsigned short* wq = wswz + (size_t)(l * 4 + 0) * 16384;
    const unsigned short* wk = wswz + (size_t)(l * 4 + 1) * 16384;
    const unsigned short* wv = wswz + (size_t)(l * 4 + 2) * 16384;
    const unsigned short* wsk= wswz + (size_t)(l * 4 + 3) * 16384;

    gemm_bf16_kernel<<<gemmBlocks, 128, 0, stream>>>(hbf, wq,  bqp, qb);
    gemm_bf16_kernel<<<gemmBlocks, 128, 0, stream>>>(hbf, wk,  bkp, kb);
    gemm_bf16_kernel<<<gemmBlocks, 128, 0, stream>>>(hbf, wv,  bvp, vb);
    gemm_bf16_kernel<<<gemmBlocks, 128, 0, stream>>>(hbf, wsk, bsp, sb);

    init_md_kernel<<<(N_NODES * HEADS + 255) / 256, 256, 0, stream>>>(
        mkeys, denom, N_NODES * HEADS);
    edge_logits_kernel<<<(ehThreads + 255) / 256, 256, 0, stream>>>(
        srcIdx, dstIdx, qb, kb, exbuf, mkeys);
    edge_exp_kernel<<<(ehThreads + 255) / 256, 256, 0, stream>>>(
        dstIdx, mkeys, exbuf, denom);
    edge_scatter_kernel<<<(ehThreads + 255) / 256, 256, 0, stream>>>(
        srcIdx, dstIdx, exbuf, denom, vb, sb);     // accumulate into skip

    relu_cvt_kernel<<<(nElem4 + 255) / 256, 256, 0, stream>>>(
        sb, hbf, (l == LAYERS - 1) ? (float*)d_out : nullptr, nElem4);
  }
}
